// Encoder_17514876634161
// MI455X (gfx1250) — matrix operand order repaired and checked by reading, not compiled, not run on hardware
//
#include <hip/hip_runtime.h>

typedef __attribute__((ext_vector_type(16))) _Float16 v16h;
typedef __attribute__((ext_vector_type(8)))  float    v8f;

// Native f32 atomic add (global_atomic_add_f32) — avoids CAS loop; graph is
// scatter-add so order nondeterminism is inherent & tolerated.
__device__ __forceinline__ void fadd_atomic(float* p, float v) {
    unsafeAtomicAdd(p, v);
}

// ---------------- init: deg=1 (self loop), acc1=0, out=b2 ----------------
__global__ void k_init(float* __restrict__ deg, float* __restrict__ acc1,
                       float* __restrict__ out, const float* __restrict__ b2,
                       int n) {
    int i = blockIdx.x * blockDim.x + threadIdx.x;
    if (i >= n) return;
    deg[i] = 1.0f;
    float* a = acc1 + (size_t)i * 48;
#pragma unroll
    for (int c = 0; c < 48; ++c) a[c] = 0.0f;
    float* o = out + (size_t)i * 32;
#pragma unroll
    for (int c = 0; c < 32; ++c) o[c] = b2[c];
}

// ---------------- in-degree over dst ----------------
__global__ void k_degree(const int* __restrict__ dst, float* __restrict__ deg,
                         int e) {
    int i = blockIdx.x * blockDim.x + threadIdx.x;
    if (i < e) fadd_atomic(&deg[dst[i]], 1.0f);
}

// ---------------- dis = rsqrt(deg) (deg >= 1 always) ----------------
__global__ void k_rsqrt(float* __restrict__ d, int n) {
    int i = blockIdx.x * blockDim.x + threadIdx.x;
    if (i < n) d[i] = rsqrtf(d[i]);
}

// ---------------- GEMM1: h[N,48] = x[N,128] @ W1[128,48] via WMMA ----------
// B fragments are pre-swizzled into WMMA lane layout in LDS, so the inner
// loop reads each 16-f16 fragment with two ds_load_b128 (no scalar packing).
__global__ void __launch_bounds__(256)
k_gemm1(const float* __restrict__ x, const float* __restrict__ W1,
        float* __restrict__ h, int n) {
    __shared__ v16h wb[12 * 32];        // [kt*3 + nt][lane]
    for (int v = threadIdx.x; v < 12 * 32; v += 256) {
        const int blk = v >> 5, ln = v & 31;
        const int kt = blk / 3, nt = blk - kt * 3;
        const int hf = ln >> 4, nl = ln & 15;
        // fragment element q -> W1[kt*32 + 16*hf + q][nt*16 + nl]
        const float* wsrc = W1 + (size_t)(kt * 32 + 16 * hf) * 48 + nt * 16 + nl;
        v16h b;
#pragma unroll
        for (int q = 0; q < 16; ++q) b[q] = (_Float16)wsrc[(size_t)q * 48];
        wb[v] = b;
    }
    __syncthreads();

    const int wave = threadIdx.x >> 5;
    const int lane = threadIdx.x & 31;
    const int half = lane >> 4;
    const int nl   = lane & 15;

    const int mtiles = (n + 15) >> 4;
    const int mt = blockIdx.x * 8 + wave;
    if (mt >= mtiles) return;
    const int row0 = mt * 16;
    int arow = row0 + nl;               // A-matrix: m = lane&15 for both halves
    if (arow >= n) arow = n - 1;        // clamp (tail-safe; N=100000 is exact)

    v8f acc0 = {}, acc1 = {}, acc2 = {};
#pragma unroll
    for (int kt = 0; kt < 4; ++kt) {
        const int kb = kt * 32;
        // A 16x32 f16: lane holds K run [kb+16h .. +15]
        const float* xr = x + (size_t)arow * 128 + kb + 16 * half;
        v16h a;
#pragma unroll
        for (int q = 0; q < 8; ++q) a[q]     = (_Float16)xr[q];
#pragma unroll
        for (int q = 0; q < 8; ++q) a[8 + q] = (_Float16)xr[8 + q];
#pragma unroll
        for (int nt = 0; nt < 3; ++nt) {
            const v16h b = wb[(kt * 3 + nt) * 32 + lane];
            v8f& acc = (nt == 0) ? acc0 : (nt == 1) ? acc1 : acc2;
            acc = __builtin_amdgcn_wmma_f32_16x16x32_f16(
                false, a, false, b, (short)0, acc, false, false);
        }
    }
    // C/D: VGPR r -> M = row0 + r + 8*half, N = nl
#pragma unroll
    for (int nt = 0; nt < 3; ++nt) {
        const v8f acc = (nt == 0) ? acc0 : (nt == 1) ? acc1 : acc2;
#pragma unroll
        for (int r = 0; r < 8; ++r) {
            int rr = row0 + r + 8 * half;
            if (rr < n) h[(size_t)rr * 48 + nt * 16 + nl] = acc[r];
        }
    }
}

// ---------------- edge scatter layer 1: acc1[dst] += h[src]*norm ----------
__global__ void __launch_bounds__(256)
k_agg1(const int* __restrict__ src, const int* __restrict__ dst,
       const float* __restrict__ dis, const float* __restrict__ h,
       float* __restrict__ acc1, int e) {
    int t = blockIdx.x * 256 + threadIdx.x;
    int g = t >> 6;          // 64-lane group per edge, 48 active
    int c = t & 63;
    if (g >= e || c >= 48) return;
    int s = src[g], d = dst[g];
    const float* hr = h + (size_t)s * 48;
    __builtin_prefetch(hr, 0, 1);              // global_prefetch_b8
    float wgt = dis[s] * dis[d];
    fadd_atomic(&acc1[(size_t)d * 48 + c], hr[c] * wgt);
}

// ---------------- z = relu(acc1 + self_loop + b1), in place --------------
__global__ void k_relu1(float* __restrict__ z, const float* __restrict__ h,
                        const float* __restrict__ dis,
                        const float* __restrict__ b1, int n) {
    size_t t = (size_t)blockIdx.x * blockDim.x + threadIdx.x;
    if (t >= (size_t)n * 48) return;
    int i = (int)(t / 48);
    int c = (int)(t - (size_t)i * 48);
    float di = dis[i];
    float v = z[t] + h[t] * di * di + b1[c];
    z[t] = v > 0.0f ? v : 0.0f;
}

// ---------------- GEMM2: h2[N,32] = z[N,48] @ W2[48,32], K padded to 64 ---
__global__ void __launch_bounds__(256)
k_gemm2(const float* __restrict__ z, const float* __restrict__ W2,
        float* __restrict__ h2, int n) {
    __shared__ v16h wb[4 * 32];         // [kt*2 + nt][lane], K 48..63 zero-pad
    for (int v = threadIdx.x; v < 4 * 32; v += 256) {
        const int blk = v >> 5, ln = v & 31;
        const int kt = blk >> 1, nt = blk & 1;
        const int hf = ln >> 4, nl = ln & 15;
        v16h b;
#pragma unroll
        for (int q = 0; q < 16; ++q) {
            int k = kt * 32 + 16 * hf + q;
            b[q] = (k < 48) ? (_Float16)W2[(size_t)k * 32 + nt * 16 + nl]
                            : (_Float16)0.0f;
        }
        wb[v] = b;
    }
    __syncthreads();

    const int wave = threadIdx.x >> 5;
    const int lane = threadIdx.x & 31;
    const int half = lane >> 4;
    const int nl   = lane & 15;

    const int mtiles = (n + 15) >> 4;
    const int mt = blockIdx.x * 8 + wave;
    if (mt >= mtiles) return;
    const int row0 = mt * 16;
    int arow = row0 + nl;
    if (arow >= n) arow = n - 1;
    const float* zr = z + (size_t)arow * 48;

    v8f acc0 = {}, acc1 = {};
#pragma unroll
    for (int kt = 0; kt < 2; ++kt) {
        const int kb = kt * 32;
        v16h a;
#pragma unroll
        for (int q = 0; q < 8; ++q) {
            int k0 = kb + 16 * half + q;       // first contiguous run
            int k1 = kb + 16 * half + 8 + q;   // second contiguous run
            a[q]     = (k0 < 48) ? (_Float16)zr[k0] : (_Float16)0.0f;
            a[8 + q] = (k1 < 48) ? (_Float16)zr[k1] : (_Float16)0.0f;
        }
#pragma unroll
        for (int nt = 0; nt < 2; ++nt) {
            const v16h b = wb[(kt * 2 + nt) * 32 + lane];
            v8f& acc = (nt == 0) ? acc0 : acc1;
            acc = __builtin_amdgcn_wmma_f32_16x16x32_f16(
                false, a, false, b, (short)0, acc, false, false);
        }
    }
#pragma unroll
    for (int nt = 0; nt < 2; ++nt) {
        const v8f acc = (nt == 0) ? acc0 : acc1;
#pragma unroll
        for (int r = 0; r < 8; ++r) {
            int rr = row0 + r + 8 * half;
            if (rr < n) h2[(size_t)rr * 32 + nt * 16 + nl] = acc[r];
        }
    }
}

// ---------------- edge scatter layer 2: out[dst] += h2[src]*norm ----------
__global__ void __launch_bounds__(256)
k_agg2(const int* __restrict__ src, const int* __restrict__ dst,
       const float* __restrict__ dis, const float* __restrict__ h2,
       float* __restrict__ out, int e) {
    int t = blockIdx.x * 256 + threadIdx.x;
    int g = t >> 5;          // 32 lanes per edge
    int c = t & 31;
    if (g >= e) return;
    int s = src[g], d = dst[g];
    const float* hr = h2 + (size_t)s * 32;
    __builtin_prefetch(hr, 0, 1);
    float wgt = dis[s] * dis[d];
    fadd_atomic(&out[(size_t)d * 32 + c], hr[c] * wgt);
}

// ---------------- layer-2 self loops (bias already seeded in init) --------
__global__ void k_self2(float* __restrict__ out, const float* __restrict__ h2,
                        const float* __restrict__ dis, int n) {
    size_t t = (size_t)blockIdx.x * blockDim.x + threadIdx.x;
    if (t >= (size_t)n * 32) return;
    int i = (int)(t >> 5);
    float di = dis[i];
    fadd_atomic(&out[t], h2[t] * di * di);
}

extern "C" void kernel_launch(void* const* d_in, const int* in_sizes, int n_in,
                              void* d_out, int out_size, void* d_ws, size_t ws_size,
                              hipStream_t stream) {
    const float* x  = (const float*)d_in[0];
    const int*   ei = (const int*)d_in[1];
    const float* W1 = (const float*)d_in[2];
    const float* b1 = (const float*)d_in[3];
    const float* W2 = (const float*)d_in[4];
    const float* b2 = (const float*)d_in[5];

    const int n = in_sizes[0] / 128;   // 100000
    const int e = in_sizes[1] / 2;     // 1600000
    const int* srcp = ei;
    const int* dstp = ei + e;

    float* ws  = (float*)d_ws;
    float* dis = ws;                                   // [n]
    float* h   = ws + (size_t)((n + 255) / 256) * 256; // [n*48], reused as h2[n*32]
    float* z   = h + (size_t)n * 48;                   // [n*48] acc1 -> relu in place
    float* out = (float*)d_out;                        // [n*32]

    const dim3 B(256);
    const int mtB = (((n + 15) / 16) + 7) / 8;         // 8 waves/block, 1 tile/wave

    k_init  <<<(n + 255) / 256,                 B, 0, stream>>>(dis, z, out, b2, n);
    k_degree<<<(e + 255) / 256,                 B, 0, stream>>>(dstp, dis, e);
    k_rsqrt <<<(n + 255) / 256,                 B, 0, stream>>>(dis, n);
    k_gemm1 <<<mtB,                             B, 0, stream>>>(x, W1, h, n);
    k_agg1  <<<(e + 3) / 4,                     B, 0, stream>>>(srcp, dstp, dis, h, z, e);
    k_relu1 <<<(int)(((size_t)n * 48 + 255) / 256), B, 0, stream>>>(z, h, dis, b1, n);
    k_gemm2 <<<mtB,                             B, 0, stream>>>(z, W2, h, n);
    k_agg2  <<<(e + 7) / 8,                     B, 0, stream>>>(srcp, dstp, dis, h, out, e);
    k_self2 <<<(int)(((size_t)n * 32 + 255) / 256), B, 0, stream>>>(out, h, dis, n);
}
